// GraphAugmentedForecastingModel_5695126635095
// MI455X (gfx1250) — compile-verified, matrix operand order checked by reading, not run on hardware
//
#include <hip/hip_runtime.h>
#include <hip/hip_bf16.h>

#define NN 5000
#define HL 64
#define GH 32
#define TT 96

typedef _Float16 v16h __attribute__((ext_vector_type(16)));
typedef _Float16 v8h  __attribute__((ext_vector_type(8)));
typedef float    v8f  __attribute__((ext_vector_type(8)));

// Fast activations. Prefer the gfx1250 native v_tanh_f32 transcendental
// (1 TRANS op, co-executes with XDL WMMA); fall back to exp+rcp.
__device__ __forceinline__ float ftanh(float z) {
#if __has_builtin(__builtin_amdgcn_tanhf)
    return __builtin_amdgcn_tanhf(z);
#else
    float e = __expf(2.0f * z);
    return 1.0f - 2.0f * __builtin_amdgcn_rcpf(e + 1.0f);
#endif
}
__device__ __forceinline__ float fsig(float z) {
#if __has_builtin(__builtin_amdgcn_tanhf)
    // sigmoid(z) = 0.5*tanh(z/2) + 0.5  : mul + tanh + fma
    return fmaf(0.5f, __builtin_amdgcn_tanhf(0.5f * z), 0.5f);
#else
    return __builtin_amdgcn_rcpf(1.0f + __expf(-z));
#endif
}

// ---------------- GCN helper kernels (tiny cost; scalar path) ----------------

__global__ void k_fill(float* __restrict__ p, float v, int n) {
    int i = blockIdx.x * blockDim.x + threadIdx.x;
    if (i < n) p[i] = v;
}

__global__ void k_deg(const int* __restrict__ dst, float* __restrict__ deg, int E) {
    int i = blockIdx.x * blockDim.x + threadIdx.x;
    if (i < E) unsafeAtomicAdd(&deg[dst[i]], 1.0f);
}

__global__ void k_dinv(const float* __restrict__ deg, float* __restrict__ dinv, int n) {
    int i = blockIdx.x * blockDim.x + threadIdx.x;
    if (i < n) dinv[i] = rsqrtf(deg[i]);
}

// out[n][32] = (relu_in ? relu(in) : in) @ W   (W is K x 32)
__global__ void k_xw(const float* __restrict__ in, const float* __restrict__ W,
                     float* __restrict__ outb, int K, int rows, int relu_in) {
    int i = blockIdx.x * blockDim.x + threadIdx.x;
    if (i >= rows * 32) return;
    int n = i >> 5, f = i & 31;
    float s = 0.0f;
    for (int k = 0; k < K; ++k) {
        float v = in[n * K + k];
        if (relu_in) v = fmaxf(v, 0.0f);
        s += v * W[k * 32 + f];
    }
    outb[i] = s;
}

// agg[n][f] = xw[n][f] * dinv[n]^2 + b[f]   (self-loop term + bias init)
__global__ void k_self(const float* __restrict__ xw, const float* __restrict__ dinv,
                       const float* __restrict__ b, float* __restrict__ agg, int rows) {
    int i = blockIdx.x * blockDim.x + threadIdx.x;
    if (i >= rows * 32) return;
    int n = i >> 5, f = i & 31;
    agg[i] = xw[i] * dinv[n] * dinv[n] + b[f];
}

// agg[dst][f] += xw[src][f] * dinv[src] * dinv[dst]
__global__ void k_scatter(const int* __restrict__ src, const int* __restrict__ dst,
                          const float* __restrict__ xw, const float* __restrict__ dinv,
                          float* __restrict__ agg, int E) {
    long i = (long)blockIdx.x * blockDim.x + threadIdx.x;
    if (i >= (long)E * 32) return;
    int e = (int)(i >> 5), f = (int)(i & 31);
    int s = src[e], d = dst[e];
    float v = xw[s * 32 + f] * dinv[s] * dinv[d];
    unsafeAtomicAdd(&agg[d * 32 + f], v);
}

// out[b*NN+n] = fc_b + sum_f relu(emb[n][f]) * fcW[64+f]  (LSTM part added later)
__global__ void k_outinit(const float* __restrict__ emb, const float* __restrict__ fcW,
                          const float* __restrict__ fcb, float* __restrict__ out, int BN) {
    int i = blockIdx.x * blockDim.x + threadIdx.x;
    if (i >= BN) return;
    int n = i % NN;
    float s = fcb[0];
    #pragma unroll
    for (int f = 0; f < 32; ++f) s += fmaxf(emb[n * 32 + f], 0.0f) * fcW[64 + f];
    out[i] = s;
}

// ---------------- LSTM: WMMA f16 recurrent GEMM ----------------
//
// Per wave: 16 sequences. Gates G = h @ W_hh^T : A = h (16x64 f16),
// B = W_hh^T (64x256 f16), done as 16 N-tiles x 2 K-chunks of
// v_wmma_f32_16x16x32_f16. Accumulators live in the C layout:
//   lane l -> gate column n = 16*tile + (l&15); VGPR j -> seq m = (l<16? j : j+8)
// Cell state c stays in VGPRs (lane-local across i/f/g/o tiles).
// h round-trips through per-wave LDS [m][k] f16; A fragments are two
// ds_load_b128 per K-chunk (A layout K-runs are 8 contiguous halfs).
// W_hh^T is pre-swizzled into fragment-contiguous LDS once per block.

#define SMEM_BYTES (32768 + 2048 + 16384 + 8192)

__global__ __launch_bounds__(256)
void k_lstm(const float* __restrict__ x, const float* __restrict__ whh,
            const float* __restrict__ wih, const float* __restrict__ bih,
            const float* __restrict__ bhh, const float* __restrict__ fcW,
            float* __restrict__ out) {
    extern __shared__ char smem[];
    _Float16* wfrag  = (_Float16*)smem;                    // 16384 halfs = 32 KB
    float*    bias_s = (float*)(smem + 32768);             // 256
    float*    wih_s  = bias_s + 256;                       // 256
    _Float16* h_s0   = (_Float16*)(wih_s + 256);           // 8 waves * 16*64 halfs
    float*    x_s0   = (float*)(h_s0 + 8 * 16 * 64);       // 8 waves * 16*16 f32

    const int tid = threadIdx.x;

    // --- Pre-swizzle W_hh^T (64x256) into B-fragment layout in LDS ---
    // fragment index = (tile*2 + chunk); per fragment: 32 lanes x 16 halfs.
    // B layout: lane l -> column n = tile*16 + (l&15);
    //           half j -> k = chunk*32 + (l&16 ? 8 : 0) + (j<8 ? j : j+8)
    for (int c4 = 0; c4 < 4; ++c4) {
        int combo = tid * 4 + c4;            // 0..1023
        int l     = combo & 31;
        int tc    = combo >> 5;              // tile*2 + chunk
        int chunk = tc & 1, tile = tc >> 1;
        int n     = tile * 16 + (l & 15);
        int kbase = chunk * 32 + ((l & 16) ? 8 : 0);
        _Float16* dp = wfrag + combo * 16;
        #pragma unroll
        for (int j = 0; j < 16; ++j) {
            int k = kbase + (j < 8 ? j : j + 8);
            dp[j] = (_Float16)whh[n * HL + k];   // W_hh^T[k][n] = W_hh[n][k]
        }
    }
    if (tid < 256) {
        bias_s[tid] = bih[tid] + bhh[tid];
        wih_s[tid]  = wih[tid];
    }
    __syncthreads();

    const int lane = tid & 31;
    const int wv   = tid >> 5;
    const long s0  = ((long)blockIdx.x * 8 + wv) * 16;   // first sequence of tile
    _Float16* h_s  = h_s0 + wv * (16 * HL);
    float*    x_s  = x_s0 + wv * (16 * 16);

    const int lh = lane & 15;
    const int hi = lane >> 4;          // 0 = lanes 0-15, 1 = lanes 16-31

    // Per-lane constants (gate column n = tile*16 + lh is step-invariant)
    float biasr[16], wihr[16];
    #pragma unroll
    for (int t = 0; t < 16; ++t) {
        int n = t * 16 + lh;
        biasr[t] = bias_s[n];
        wihr[t]  = wih_s[n];
    }
    float fcr[4];
    #pragma unroll
    for (int t = 0; t < 4; ++t) fcr[t] = fcW[t * 16 + lh];

    // h, c = 0
    for (int i = lane; i < 16 * HL; i += 32) h_s[i] = (_Float16)0.0f;
    float cst[4][8];
    #pragma unroll
    for (int t = 0; t < 4; ++t)
        #pragma unroll
        for (int j = 0; j < 8; ++j) cst[t][j] = 0.0f;

    float dot[8];
    #pragma unroll
    for (int j = 0; j < 8; ++j) dot[j] = 0.0f;

    union AF { v16h v; v8h h[2]; };

    for (int step = 0; step < TT; ++step) {
        const int ch = step & 15;
        if (ch == 0) {
            // stage 16-step x chunk for this wave's 16 sequences (coalesced)
            const int cb = step >> 4;
            #pragma unroll
            for (int r = 0; r < 8; ++r) {
                int q = r * 32 + lane;
                int m = q >> 4, t2 = q & 15;
                x_s[q] = x[(s0 + m) * TT + cb * 16 + t2];
            }
            if (cb < 5)   // hint next chunk into cache (global_prefetch_b8)
                __builtin_prefetch(x + (s0 + (lane & 15)) * TT + (cb + 1) * 16, 0, 1);
        }

        // x for this step, per accumulator row m
        float xv[8];
        #pragma unroll
        for (int j = 0; j < 8; ++j) {
            int m = hi ? (j + 8) : j;
            xv[j] = x_s[m * 16 + ch];
        }

        // A fragments (h from previous step), two ds_load_b128 per K-chunk
        AF a0, a1;
        {
            const _Float16* hp = h_s + lh * HL + (hi ? 8 : 0);
            a0.h[0] = *(const v8h*)(hp);        // K = base .. base+7
            a0.h[1] = *(const v8h*)(hp + 16);   // K = base+16 .. base+23
            a1.h[0] = *(const v8h*)(hp + 32);   // chunk 1
            a1.h[1] = *(const v8h*)(hp + 48);
        }

        #pragma unroll
        for (int t = 0; t < 4; ++t) {           // hidden chunk: k = 16t + lh
            v8f acc[4];
            #pragma unroll
            for (int g = 0; g < 4; ++g) {       // i, f, g, o
                const int tile = g * 4 + t;
                const float b0 = biasr[tile], w0 = wihr[tile];
                #pragma unroll
                for (int j = 0; j < 8; ++j) acc[g][j] = b0 + w0 * xv[j];
            }
            #pragma unroll
            for (int g = 0; g < 4; ++g) {
                const int tile = g * 4 + t;
                const _Float16* bp0 = wfrag + ((tile * 2 + 0) * 32 + lane) * 16;
                const _Float16* bp1 = wfrag + ((tile * 2 + 1) * 32 + lane) * 16;
                AF b0f, b1f;
                b0f.h[0] = *(const v8h*)(bp0);
                b0f.h[1] = *(const v8h*)(bp0 + 8);
                b1f.h[0] = *(const v8h*)(bp1);
                b1f.h[1] = *(const v8h*)(bp1 + 8);
                acc[g] = __builtin_amdgcn_wmma_f32_16x16x32_f16(
                    false, a0.v, false, b0f.v, (short)0, acc[g], false, false);
                acc[g] = __builtin_amdgcn_wmma_f32_16x16x32_f16(
                    false, a1.v, false, b1f.v, (short)0, acc[g], false, false);
            }
            // lane-local cell update; store h back to LDS in [m][k] f16
            const int kt = t * 16 + lh;
            #pragma unroll
            for (int j = 0; j < 8; ++j) {
                float ig = fsig(acc[0][j]);
                float fg = fsig(acc[1][j]);
                float gg = ftanh(acc[2][j]);
                float og = fsig(acc[3][j]);
                float cc = fg * cst[t][j] + ig * gg;
                cst[t][j] = cc;
                float hh = og * ftanh(cc);
                int m = hi ? (j + 8) : j;
                h_s[m * HL + kt] = (_Float16)hh;
                if (step == TT - 1) dot[j] += hh * fcr[t];
            }
        }
    }

    // reduce h_last . fcW[0:64] over the 16 lanes of each half (xor masks stay in-half)
    #pragma unroll
    for (int mk = 1; mk <= 8; mk <<= 1) {
        #pragma unroll
        for (int j = 0; j < 8; ++j) dot[j] += __shfl_xor(dot[j], mk, 32);
    }
    if (lh == 0) {
        #pragma unroll
        for (int j = 0; j < 8; ++j) {
            long m = s0 + (hi ? j + 8 : j);
            out[m] += dot[j];   // single writer per m; out pre-initialized
        }
    }
}

// ---------------- launch ----------------

extern "C" void kernel_launch(void* const* d_in, const int* in_sizes, int n_in,
                              void* d_out, int out_size, void* d_ws, size_t ws_size,
                              hipStream_t stream) {
    const float* x    = (const float*)d_in[0];
    const float* nf   = (const float*)d_in[1];
    const int*   ei   = (const int*)d_in[2];
    const float* W1   = (const float*)d_in[3];
    const float* b1   = (const float*)d_in[4];
    const float* W2   = (const float*)d_in[5];
    const float* b2   = (const float*)d_in[6];
    const float* Wih  = (const float*)d_in[7];
    const float* Whh  = (const float*)d_in[8];
    const float* bih  = (const float*)d_in[9];
    const float* bhh  = (const float*)d_in[10];
    const float* fcW  = (const float*)d_in[11];
    const float* fcb  = (const float*)d_in[12];
    float* out = (float*)d_out;

    const int E  = in_sizes[2] / 2;
    const int* src = ei;
    const int* dst = ei + E;
    const int B  = in_sizes[0] / (NN * TT);
    const int BN = B * NN;

    float* ws   = (float*)d_ws;
    float* deg  = ws;                 // 5000
    float* dinv = deg + 5120;
    float* xw1  = dinv + 5120;        // 5000*32
    float* agg1 = xw1 + NN * 32;
    float* xw2  = agg1 + NN * 32;
    float* agg2 = xw2 + NN * 32;

    // --- GCN (tiny) ---
    k_fill<<<(NN + 255) / 256, 256, 0, stream>>>(deg, 1.0f, NN);       // self-loop
    k_deg<<<(E + 255) / 256, 256, 0, stream>>>(dst, deg, E);
    k_dinv<<<(NN + 255) / 256, 256, 0, stream>>>(deg, dinv, NN);

    k_xw<<<(NN * 32 + 255) / 256, 256, 0, stream>>>(nf, W1, xw1, 16, NN, 0);
    k_self<<<(NN * 32 + 255) / 256, 256, 0, stream>>>(xw1, dinv, b1, agg1, NN);
    k_scatter<<<(int)(((long)E * 32 + 255) / 256), 256, 0, stream>>>(src, dst, xw1, dinv, agg1, E);

    k_xw<<<(NN * 32 + 255) / 256, 256, 0, stream>>>(agg1, W2, xw2, 32, NN, 1); // relu folded
    k_self<<<(NN * 32 + 255) / 256, 256, 0, stream>>>(xw2, dinv, b2, agg2, NN);
    k_scatter<<<(int)(((long)E * 32 + 255) / 256), 256, 0, stream>>>(src, dst, xw2, dinv, agg2, E);

    // out = fc_b + relu(emb) . fcW[64:96]
    k_outinit<<<(BN + 255) / 256, 256, 0, stream>>>(agg2, fcW, fcb, out, BN);

    // --- LSTM (dominant; WMMA) adds h_last . fcW[0:64] ---
    const int blocks = BN / 128;      // 8 waves/block * 16 seqs/wave
    k_lstm<<<blocks, 256, SMEM_BYTES, stream>>>(x, Whh, Wih, bih, bhh, fcW, out);
}
